// NeuralCellularAutomata2_34299608825955
// MI455X (gfx1250) — matrix operand order repaired and checked by reading, not compiled, not run on hardware
//
#include <hip/hip_runtime.h>

typedef __attribute__((ext_vector_type(16))) _Float16 v16h;
typedef __attribute__((ext_vector_type(8)))  _Float16 v8h;
typedef __attribute__((ext_vector_type(4)))  _Float16 v4h;
typedef __attribute__((ext_vector_type(8)))  float    v8f;

#define BATCH 8
#define HH 128
#define WW 128
#define C 64          // hidden
#define C3 192        // 3C
#define C2 128        // 2C

// tile: 4 rows x 16 cols interior, 4 waves of 32 (128 threads)
#define TR 4
#define TC 16

__device__ __forceinline__ v16h load_frag(const _Float16* p0, const _Float16* p1) {
  v8h a = *(const v8h*)p0;
  v8h b = *(const v8h*)p1;
  return __builtin_shufflevector(a, b, 0,1,2,3,4,5,6,7,8,9,10,11,12,13,14,15);
}

__device__ __forceinline__ v8f wmma_f16(v16h a, v16h b, v8f c) {
  return __builtin_amdgcn_wmma_f32_16x16x32_f16(false, a, false, b, (short)0, c, false, false);
}

// async copy one dword global -> LDS (ASYNCcnt path); LDS offset = low 32 bits
// of the generic pointer.
__device__ __forceinline__ void async_copy_b32(float* lds_dst, const float* gsrc) {
  unsigned loff = (unsigned)(unsigned long long)lds_dst;
  asm volatile("global_load_async_to_lds_b32 %0, %1, off"
               :: "v"(loff), "v"((unsigned long long)gsrc) : "memory");
}
__device__ __forceinline__ void async_wait0() {
  asm volatile("s_wait_asynccnt 0" ::: "memory");
}

// ---------------- kernel 0: pack weights into f16 WMMA B-fragment order -------
// packed[((nb*KB + kb)*32 + L)*16 + j] = W[nb*16 + (L&15)][kb*32 + 8*(L>>4) + (j&7) + 16*(j>>3)]
__global__ void pack_weights(const float* __restrict__ w1, const float* __restrict__ w2,
                             const float* __restrict__ wqkv,
                             _Float16* __restrict__ pw1, _Float16* __restrict__ pw2,
                             _Float16* __restrict__ pwqkv) {
  int t = blockIdx.x * 256 + threadIdx.x;
  if (t >= 45056) return;
  if (t < 24576) {               // w1: [128][192], KB=6, NB=8
    int rel = t, tile = rel >> 9, within = rel & 511;
    int L = within >> 4, j = within & 15;
    int kb = tile % 6, nb = tile / 6;
    int n = nb * 16 + (L & 15);
    int k = kb * 32 + ((L >> 4) << 3) + (j & 7) + ((j >> 3) << 4);
    pw1[rel] = (_Float16)w1[n * C3 + k];
  } else if (t < 32768) {        // w2: [64][128], KB=4, NB=4
    int rel = t - 24576, tile = rel >> 9, within = rel & 511;
    int L = within >> 4, j = within & 15;
    int kb = tile % 4, nb = tile / 4;
    int n = nb * 16 + (L & 15);
    int k = kb * 32 + ((L >> 4) << 3) + (j & 7) + ((j >> 3) << 4);
    pw2[rel] = (_Float16)w2[n * C2 + k];
  } else {                       // wqkv: [192][64], KB=2, NB=12
    int rel = t - 32768, tile = rel >> 9, within = rel & 511;
    int L = within >> 4, j = within & 15;
    int kb = tile % 2, nb = tile / 2;
    int n = nb * 16 + (L & 15);
    int k = kb * 32 + ((L >> 4) << 3) + (j & 7) + ((j >> 3) << 4);
    pwqkv[rel] = (_Float16)wqkv[n * C + k];
  }
}

// ---------------- kernel 1: depthwise conv + update_net (2 WMMA GEMMs) --------
__global__ __launch_bounds__(128) void nca_update(
    const float* __restrict__ h, const float* __restrict__ wperc,
    const float* __restrict__ bperc, const float* __restrict__ b1,
    const float* __restrict__ b2, const _Float16* __restrict__ pw1,
    const _Float16* __restrict__ pw2, float* __restrict__ hnew) {
  __shared__ __align__(16) _Float16 sh_h[6 * 18 * C];      // halo tile of h (f16)
  __shared__ __align__(16) _Float16 sh_perc[64 * C3];      // perceived
  __shared__ __align__(16) _Float16 sh_mid[64 * C2];       // GELU(mid)
  __shared__ __align__(16) float sh_wp[C3 * 9];

  const int tid = threadIdx.x;
  const int bi = blockIdx.x;
  const int b  = bi >> 8;                  // 32*8 tiles per image
  const int tr = (bi >> 3) & 31;
  const int tc = bi & 7;
  const int r0 = tr * TR, c0 = tc * TC;

  // async DMA the depthwise weight table straight into LDS (no VGPR staging)
  for (int i = tid; i < C3 * 9; i += 128) async_copy_b32(&sh_wp[i], &wperc[i]);

  // prefetch packed GEMM weights into cache while we do the conv
  __builtin_prefetch(pw1 + (tid << 5), 0, 0);
  __builtin_prefetch(pw2 + (tid << 5), 0, 0);

  // load 6x18 halo of h (zero padded), 4 channels per thread (b128 loads)
  for (int i = tid; i < 6 * 18 * 16; i += 128) {
    int pix = i >> 4, c4 = (i & 15) << 2;
    int hr = pix / 18, hc = pix % 18;
    int gr = r0 + hr - 1, gc = c0 + hc - 1;
    float4 v = make_float4(0.f, 0.f, 0.f, 0.f);
    if (gr >= 0 && gr < HH && gc >= 0 && gc < WW)
      v = *(const float4*)(h + (((long)b * HH + gr) * WW + gc) * C + c4);
    *(v4h*)(&sh_h[(pix << 6) + c4]) =
        (v4h){(_Float16)v.x, (_Float16)v.y, (_Float16)v.z, (_Float16)v.w};
  }
  async_wait0();
  __syncthreads();

  // depthwise 3x3, groups=C, 3 out channels per group: in channel = oc/3
  for (int i = tid; i < 64 * C3; i += 128) {
    int oc = i % C3, p = i / C3;
    int pr = p >> 4, pc = p & 15;
    int cin = oc / 3;
    float acc = bperc[oc];
#pragma unroll
    for (int kh = 0; kh < 3; ++kh)
#pragma unroll
      for (int kw = 0; kw < 3; ++kw)
        acc += (float)sh_h[((pr + kh) * 18 + (pc + kw)) * C + cin] *
               sh_wp[oc * 9 + kh * 3 + kw];
    sh_perc[p * C3 + oc] = (_Float16)acc;
  }
  __syncthreads();

  const int wave = tid >> 5, lane = tid & 31;
  const int mrow = wave * 16 + (lane & 15);
  const int koff = (lane >> 4) * 8;

  // GEMM1: [16x192] x [192x128] + b1, then exact GELU
  v8f acc[8];
#pragma unroll
  for (int nb = 0; nb < 8; ++nb) {
    float bv = b1[nb * 16 + (lane & 15)];
#pragma unroll
    for (int r = 0; r < 8; ++r) acc[nb][r] = bv;
  }
#pragma unroll
  for (int kb = 0; kb < 6; ++kb) {
    const _Float16* ap = sh_perc + mrow * C3 + kb * 32 + koff;
    v16h a = load_frag(ap, ap + 16);
#pragma unroll
    for (int nb = 0; nb < 8; ++nb) {
      const _Float16* bp = pw1 + (((nb * 6 + kb) * 32 + lane) << 4);
      v16h bf = load_frag(bp, bp + 8);
      acc[nb] = wmma_f16(a, bf, acc[nb]);
    }
  }
#pragma unroll
  for (int nb = 0; nb < 8; ++nb)
#pragma unroll
    for (int r = 0; r < 8; ++r) {
      float x = acc[nb][r];
      float g = 0.5f * x * (1.0f + erff(x * 0.70710678118654752f));
      int mr = wave * 16 + r + ((lane >> 4) << 3);  // C-matrix row
      sh_mid[mr * C2 + nb * 16 + (lane & 15)] = (_Float16)g;
    }
  __syncthreads();

  // GEMM2: [16x128] x [128x64] + b2, residual add with h
  v8f acc2[4];
#pragma unroll
  for (int nb = 0; nb < 4; ++nb) {
    float bv = b2[nb * 16 + (lane & 15)];
#pragma unroll
    for (int r = 0; r < 8; ++r) acc2[nb][r] = bv;
  }
#pragma unroll
  for (int kb = 0; kb < 4; ++kb) {
    const _Float16* ap = sh_mid + mrow * C2 + kb * 32 + koff;
    v16h a = load_frag(ap, ap + 16);
#pragma unroll
    for (int nb = 0; nb < 4; ++nb) {
      const _Float16* bp = pw2 + (((nb * 4 + kb) * 32 + lane) << 4);
      v16h bf = load_frag(bp, bp + 8);
      acc2[nb] = wmma_f16(a, bf, acc2[nb]);
    }
  }
#pragma unroll
  for (int nb = 0; nb < 4; ++nb)
#pragma unroll
    for (int r = 0; r < 8; ++r) {
      int mloc = r + ((lane >> 4) << 3);     // col within tile row
      int gr = r0 + wave, gc = c0 + mloc;
      int oc = nb * 16 + (lane & 15);
      long idx = (((long)b * HH + gr) * WW + gc) * C + oc;
      hnew[idx] = acc2[nb][r] + h[idx];
    }
}

// ---------------- kernel 2: QKV (WMMA) + 3x3 windowed attention ---------------
__global__ __launch_bounds__(128) void qkv_attn(
    const float* __restrict__ hnew, const float* __restrict__ bqkv,
    const _Float16* __restrict__ pwqkv, float* __restrict__ out) {
  __shared__ __align__(16) _Float16 sh_a[112 * C];       // h_new halo (108 + pad)
  __shared__ __align__(16) _Float16 sh_qkv[108 * C3];

  const int tid = threadIdx.x;
  const int bi = blockIdx.x;
  const int b  = bi >> 8;
  const int tr = (bi >> 3) & 31;
  const int tc = bi & 7;
  const int r0 = tr * TR, c0 = tc * TC;

  __builtin_prefetch(pwqkv + (tid << 5), 0, 0);

  // 108 halo rows, 4 channels/thread; rows 108..111 zero pad
  for (int i = tid; i < 108 * 16; i += 128) {
    int row = i >> 4, c4 = (i & 15) << 2;
    int hr = row / 18, hc = row % 18;
    int gr = r0 + hr - 1, gc = c0 + hc - 1;
    float4 v = make_float4(0.f, 0.f, 0.f, 0.f);
    if (gr >= 0 && gr < HH && gc >= 0 && gc < WW)
      v = *(const float4*)(hnew + (((long)b * HH + gr) * WW + gc) * C + c4);
    *(v4h*)(&sh_a[(row << 6) + c4]) =
        (v4h){(_Float16)v.x, (_Float16)v.y, (_Float16)v.z, (_Float16)v.w};
  }
  for (int i = tid; i < 4 * C; i += 128) sh_a[108 * C + i] = (_Float16)0.f;
  __syncthreads();

  const int wave = tid >> 5, lane = tid & 31;
  const int koff = (lane >> 4) * 8;

  // qkv for all 108 halo pixels: 7 M-tiles of 16, K=64 (2 blocks), N=192 (12)
  for (int mt = wave; mt < 7; mt += 4) {
    int mrow = mt * 16 + (lane & 15);
    const _Float16* ap0 = sh_a + mrow * C + koff;
    const _Float16* ap1 = sh_a + mrow * C + 32 + koff;
    v16h a0 = load_frag(ap0, ap0 + 16);
    v16h a1 = load_frag(ap1, ap1 + 16);
#pragma unroll
    for (int nb = 0; nb < 12; ++nb) {
      v8f acc;
      float bv = bqkv[nb * 16 + (lane & 15)];
#pragma unroll
      for (int r = 0; r < 8; ++r) acc[r] = bv;
      const _Float16* bp0 = pwqkv + (((nb * 2 + 0) * 32 + lane) << 4);
      const _Float16* bp1 = pwqkv + (((nb * 2 + 1) * 32 + lane) << 4);
      acc = wmma_f16(a0, load_frag(bp0, bp0 + 8), acc);
      acc = wmma_f16(a1, load_frag(bp1, bp1 + 8), acc);
#pragma unroll
      for (int r = 0; r < 8; ++r) {
        int mr = mt * 16 + r + ((lane >> 4) << 3);
        if (mr < 108)
          sh_qkv[mr * C3 + nb * 16 + (lane & 15)] = (_Float16)acc[r];
      }
    }
  }
  __syncthreads();

  // attention: 2 lanes per pixel, each handles 32 channels
  const int p = tid >> 1, half = tid & 1;
  const int pr = p >> 4, pc = p & 15;
  const int gr = r0 + pr, gc = c0 + pc;
  const int hpc = (pr + 1) * 18 + (pc + 1);

  float q[32];
#pragma unroll
  for (int c = 0; c < 32; ++c)
    q[c] = (float)sh_qkv[hpc * C3 + half * 32 + c];

  int   hpn[9];
  float att[9];
#pragma unroll
  for (int n = 0; n < 9; ++n) {
    int i = n / 3, j = n % 3;
    int nr = gr + i - 1, nc = gc + j - 1;
    bool valid = (nr >= 0 && nr < HH && nc >= 0 && nc < WW);
    hpn[n] = valid ? ((pr + i) * 18 + (pc + j)) : -1;
    float s = 0.f;
    if (valid) {
      const _Float16* kp = sh_qkv + hpn[n] * C3 + 64 + half * 32;
      float part = 0.f;
#pragma unroll
      for (int c = 0; c < 32; ++c) part += q[c] * (float)kp[c];
      s = (part + __shfl_xor(part, 1, 32)) * 0.125f;   // / sqrt(64)
    }
    att[n] = s;                  // zero-padded K => logit exactly 0 for OOB
  }
  float mx = att[0];
#pragma unroll
  for (int n = 1; n < 9; ++n) mx = fmaxf(mx, att[n]);
  float den = 0.f;
#pragma unroll
  for (int n = 0; n < 9; ++n) { att[n] = __expf(att[n] - mx); den += att[n]; }
  float inv = 1.0f / den;

#pragma unroll
  for (int c = 0; c < 32; ++c) {
    int ch = half * 32 + c;
    float o = 0.f;
#pragma unroll
    for (int n = 0; n < 9; ++n)
      if (hpn[n] >= 0)
        o += att[n] * (float)sh_qkv[hpn[n] * C3 + 128 + ch];
    long idx = (((long)b * HH + gr) * WW + gc) * C + ch;
    out[idx] = hnew[idx] + o * inv;
  }
}

extern "C" void kernel_launch(void* const* d_in, const int* in_sizes, int n_in,
                              void* d_out, int out_size, void* d_ws, size_t ws_size,
                              hipStream_t stream) {
  const float* h     = (const float*)d_in[0];
  const float* wperc = (const float*)d_in[1];
  const float* bperc = (const float*)d_in[2];
  const float* w1    = (const float*)d_in[3];
  const float* b1    = (const float*)d_in[4];
  const float* w2    = (const float*)d_in[5];
  const float* b2    = (const float*)d_in[6];
  const float* wqkv  = (const float*)d_in[7];
  const float* bqkv  = (const float*)d_in[8];
  float* out = (float*)d_out;

  char* ws = (char*)d_ws;
  float*    hnew  = (float*)ws;                              // 33,554,432 B
  _Float16* pw1   = (_Float16*)(ws + 33554432);              // 49,152 B
  _Float16* pw2   = (_Float16*)(ws + 33554432 + 49152);      // 16,384 B
  _Float16* pwqkv = (_Float16*)(ws + 33554432 + 49152 + 16384); // 24,576 B

  pack_weights<<<176, 256, 0, stream>>>(w1, w2, wqkv, pw1, pw2, pwqkv);

  dim3 grid(BATCH * (HH / TR) * (WW / TC));   // 2048
  nca_update<<<grid, 128, 0, stream>>>(h, wperc, bperc, b1, b2, pw1, pw2, hnew);
  qkv_attn<<<grid, 128, 0, stream>>>(hnew, bqkv, pwqkv, out);
}
